// SpiralAutoencoder_71734543778286
// MI455X (gfx1250) — matrix operand order repaired and checked by reading, not compiled, not run on hardware
//
#include <hip/hip_runtime.h>
#include <math.h>

#define BATCH  64
#define SPIRAL 9
#define MT     4   // M-tiles per wave (shared B fragment)
#define WV     4   // waves per block (shared LDS-staged B tile)

typedef __attribute__((ext_vector_type(16))) __bf16 v16bf;
typedef __attribute__((ext_vector_type(8)))  float  v8f;

union Frag {
  v16bf v;
  unsigned short us[16];
  unsigned int   ui[8];
};

__device__ __forceinline__ unsigned short f32_to_bf16_rne(float f) {
  unsigned int u = __float_as_uint(f);
  u += 0x7FFFu + ((u >> 16) & 1u);
  return (unsigned short)(u >> 16);
}
__device__ __forceinline__ float bf16_to_f32(unsigned short h) {
  return __uint_as_float(((unsigned int)h) << 16);
}

// ---------------------------------------------------------------------------
// elementwise f32 -> bf16
// ---------------------------------------------------------------------------
__global__ void cvt_f32_bf16_kernel(const float* __restrict__ src,
                                    unsigned short* __restrict__ dst, int n) {
  int i = blockIdx.x * blockDim.x + threadIdx.x;
  if (i < n) dst[i] = f32_to_bf16_rne(src[i]);
}

// ---------------------------------------------------------------------------
// Fragment layouts (CDNA5 16-bit WMMA, ISA 7.12.2)
// A 16x32: lane holds row m=lane&15; VGPR j pair K = (lane>>4)*16 + 2*j (contiguous order used here, matching B)
// B 32x16: lane holds col n=lane&15; VGPR j pair K = (lane>>4)*16 + 2*j (contiguous order used here; ISA table order differs)
// C/D 16x16 f32: col n=lane&15, VGPR i row m = i + (lane>>4)*8
//
// LDS B tile is staged pre-swizzled into fragment order:
//   B[k][n]  ->  lds[((k&16)+n)*16 + (k&15)]
// so lane L's fragment is the 16 contiguous shorts at lds[L*16].
// ---------------------------------------------------------------------------
__device__ __forceinline__ v16bf load_a_f32_row(const float* __restrict__ arow, bool valid,
                                                int K, int k0, int lane) {
  Frag f;
  int base = ((lane >> 4) << 3);
#pragma unroll
  for (int j = 0; j < 8; ++j) {
    int k = k0 + (base << 1) + (j << 1);
    float x0 = 0.f, x1 = 0.f;
    if (valid) {
      if (k     < K) x0 = arow[k];
      if (k + 1 < K) x1 = arow[k + 1];
    }
    f.us[2 * j]     = f32_to_bf16_rne(x0);
    f.us[2 * j + 1] = f32_to_bf16_rne(x1);
  }
  return f.v;
}

__device__ __forceinline__ v16bf load_a_bf16_row(const unsigned short* __restrict__ arow,
                                                 int K, int k0, int lane) {
  Frag f;
  int base = ((lane >> 4) << 3);
#pragma unroll
  for (int j = 0; j < 8; ++j) {
    int k = k0 + (base << 1) + (j << 1);
    unsigned int v = 0;
    if (k + 1 < K) v = *(const unsigned int*)(arow + k);   // 2 contiguous bf16
    else if (k < K) v = (unsigned int)arow[k];
    f.ui[j] = v;
  }
  return f.v;
}

// direct-global B fragment (used by the small FC kernel)
__device__ __forceinline__ v16bf load_b_f32(const float* __restrict__ Bm, int ldb,
                                            int K, int k0, int n, int lane) {
  Frag f;
  int kb = k0 + ((lane >> 4) << 4);
#pragma unroll
  for (int j = 0; j < 8; ++j) {
    int k = kb + 2 * j;
    float x0 = (k     < K) ? Bm[(size_t)k * ldb + n]       : 0.f;
    float x1 = (k + 1 < K) ? Bm[(size_t)(k + 1) * ldb + n] : 0.f;
    f.us[2 * j]     = f32_to_bf16_rne(x0);
    f.us[2 * j + 1] = f32_to_bf16_rne(x1);
  }
  return f.v;
}

// ---------------------------------------------------------------------------
// Spiral conv via WMMA: gather fused into A-fragment load; W tile staged
// through LDS in fragment order, shared by 4 waves x 4 M-tiles.
// H: bf16 [B,Pn,Cin]; S: int [Pn,9]; W: f32 [9*Cin,Cout]; Out: bf16 [B,Pn,Cout]
// grid: (ceil(B*Pn/256), Cout/16)
// ---------------------------------------------------------------------------
__global__ __launch_bounds__(128)
void sconv_wmma_kernel(const unsigned short* __restrict__ H, const int* __restrict__ S,
                       const float* __restrict__ W, const float* __restrict__ bias,
                       unsigned short* __restrict__ Out,
                       int Pn, int cinShift, int Cout, int Kdim) {
  __shared__ __align__(32) unsigned short ldsB[512];   // 32x16 bf16 tile, frag order
  const int tid   = threadIdx.x;
  const int lane  = tid & 31;
  const int wv    = tid >> 5;
  const int Cin   = 1 << cinShift;
  const int n0    = blockIdx.y * 16;
  const int n     = n0 + (lane & 15);
  const int M     = BATCH * Pn;
  const int m0    = blockIdx.x * (16 * MT * WV) + wv * (16 * MT);
  const int abase = ((lane >> 4) << 3);

  const int* sp[MT];
  const unsigned short* hb[MT];
#pragma unroll
  for (int t = 0; t < MT; ++t) {
    int m  = m0 + t * 16 + (lane & 15);
    int mc = (m < M) ? m : (M - 1);                    // clamp for safe gather addresses
    int b  = mc / Pn;
    int p  = mc - b * Pn;
    sp[t] = S + p * SPIRAL;
    hb[t] = H + (size_t)b * Pn * Cin;
  }

  // staging indices: thread stages 4 cols of one k-row
  const int sk = tid >> 2;
  const int sn = (tid & 3) * 4;
  const int sbase = ((sk & 16) + sn) * 16 + (sk & 15);

  v8f acc[MT] = {};
  for (int k0 = 0; k0 < Kdim; k0 += 32) {
    __syncthreads();
    {
      int kg = k0 + sk;
      unsigned short v0 = 0, v1 = 0, v2 = 0, v3 = 0;
      if (kg < Kdim) {
        const float4 wv4 = *(const float4*)(W + (size_t)kg * Cout + n0 + sn);
        v0 = f32_to_bf16_rne(wv4.x); v1 = f32_to_bf16_rne(wv4.y);
        v2 = f32_to_bf16_rne(wv4.z); v3 = f32_to_bf16_rne(wv4.w);
      }
      ldsB[sbase]      = v0;
      ldsB[sbase + 16] = v1;
      ldsB[sbase + 32] = v2;
      ldsB[sbase + 48] = v3;
    }
    __syncthreads();
    Frag fb;
    fb.v = *(const v16bf*)&ldsB[lane * 16];            // contiguous frag -> ds_load_b128 x2
#pragma unroll
    for (int t = 0; t < MT; ++t) {
      Frag fa;
#pragma unroll
      for (int j = 0; j < 8; ++j) {
        int k = k0 + (abase << 1) + (j << 1);
        unsigned int v = 0;
        if (k < Kdim) {
          int s = k >> cinShift;
          int c = k & (Cin - 1);                       // even -> dword-aligned pair
          v = *(const unsigned int*)(hb[t] + (size_t)sp[t][s] * Cin + c);
        }
        fa.ui[j] = v;
      }
      acc[t] = __builtin_amdgcn_wmma_f32_16x16x32_bf16(false, fa.v, false, fb.v,
                                                       (short)0, acc[t], false, false);
    }
  }

  const float bn = bias[n];
#pragma unroll
  for (int t = 0; t < MT; ++t) {
    int mb = m0 + t * 16 + ((lane >> 4) << 3);
#pragma unroll
    for (int i = 0; i < 8; ++i) {
      int m2 = mb + i;
      if (m2 < M) {
        int b2 = m2 / Pn;
        int p2 = m2 - b2 * Pn;
        float y = acc[t][i] + bn;
        y = (y > 0.f) ? y : (__expf(y) - 1.f);         // ELU, alpha=1
        if (p2 == Pn - 1) y = 0.f;                     // zero padding vertex
        Out[((size_t)b2 * Pn + p2) * Cout + n] = f32_to_bf16_rne(y);
      }
    }
  }
}

// ---------------------------------------------------------------------------
// Pool/unpool: Out[b] = Mx[Pout,Pin](f32) x Hin[b][Pin,F](bf16) -> bf16
// B (activation) tile staged through LDS, shared by 4 waves x 4 M-tiles.
// grid: (ceil(Pout/256), F/16, BATCH)
// ---------------------------------------------------------------------------
__global__ __launch_bounds__(128)
void vertmix_wmma_kernel(const float* __restrict__ Mx, const unsigned short* __restrict__ Hin,
                         unsigned short* __restrict__ Out, int Pout, int Pin, int F) {
  __shared__ __align__(32) unsigned short ldsB[512];
  const int tid  = threadIdx.x;
  const int lane = tid & 31;
  const int wv   = tid >> 5;
  const int bz   = blockIdx.z;
  const int n0   = blockIdx.y * 16;
  const int n    = n0 + (lane & 15);
  const int m0   = blockIdx.x * (16 * MT * WV) + wv * (16 * MT);
  const unsigned short* Bm = Hin + (size_t)bz * Pin * F;

  bool av[MT];
  const float* arow[MT];
#pragma unroll
  for (int t = 0; t < MT; ++t) {
    int m = m0 + t * 16 + (lane & 15);
    av[t]   = (m < Pout);
    arow[t] = Mx + (size_t)(av[t] ? m : 0) * Pin;
  }

  const int sk = tid >> 2;
  const int sn = (tid & 3) * 4;
  const int sbase = ((sk & 16) + sn) * 16 + (sk & 15);

  v8f acc[MT] = {};
  for (int k0 = 0; k0 < Pin; k0 += 32) {
    __syncthreads();
    {
      int kg = k0 + sk;
      unsigned int d0 = 0, d1 = 0;
      if (kg < Pin) {
        const unsigned int* s = (const unsigned int*)(Bm + (size_t)kg * F + n0 + sn);
        d0 = s[0]; d1 = s[1];                          // 4 coalesced bf16
      }
      ldsB[sbase]      = (unsigned short)(d0 & 0xFFFFu);
      ldsB[sbase + 16] = (unsigned short)(d0 >> 16);
      ldsB[sbase + 32] = (unsigned short)(d1 & 0xFFFFu);
      ldsB[sbase + 48] = (unsigned short)(d1 >> 16);
    }
    __syncthreads();
    Frag fb;
    fb.v = *(const v16bf*)&ldsB[lane * 16];
#pragma unroll
    for (int t = 0; t < MT; ++t)                       // pull next A chunk toward WGP
      __builtin_prefetch(arow[t] + k0 + 128, 0, 0);    // -> global_prefetch_b8
#pragma unroll
    for (int t = 0; t < MT; ++t) {
      v16bf fa = load_a_f32_row(arow[t], av[t], Pin, k0, lane);
      acc[t] = __builtin_amdgcn_wmma_f32_16x16x32_bf16(false, fa, false, fb.v,
                                                       (short)0, acc[t], false, false);
    }
  }

#pragma unroll
  for (int t = 0; t < MT; ++t) {
    int mb = m0 + t * 16 + ((lane >> 4) << 3);
#pragma unroll
    for (int i = 0; i < 8; ++i) {
      int m2 = mb + i;
      if (m2 < Pout)
        Out[((size_t)bz * Pout + m2) * F + n] = f32_to_bf16_rne(acc[t][i]);
    }
  }
}

// ---------------------------------------------------------------------------
// FC: Out = A[64,K](bf16) x W[K,N](f32) + bias ; optional f32 copy (latent z)
// M=64 handled as 4 M-tiles in one wave; grid: (1, N/16); K multiple of 32
// ---------------------------------------------------------------------------
__global__ __launch_bounds__(32)
void fc_wmma_kernel(const unsigned short* __restrict__ A, const float* __restrict__ W,
                    const float* __restrict__ bias,
                    unsigned short* __restrict__ OutBf, float* __restrict__ OutF32,
                    int K, int N) {
  const int lane = threadIdx.x;
  const int n    = blockIdx.y * 16 + (lane & 15);

  const unsigned short* arow[MT];
#pragma unroll
  for (int t = 0; t < MT; ++t)
    arow[t] = A + (size_t)(t * 16 + (lane & 15)) * K;

  v8f acc[MT] = {};
  for (int k0 = 0; k0 < K; k0 += 32) {
    v16bf fb = load_b_f32(W, N, K, k0, n, lane);
#pragma unroll
    for (int t = 0; t < MT; ++t) {
      v16bf fa = load_a_bf16_row(arow[t], K, k0, lane);
      acc[t] = __builtin_amdgcn_wmma_f32_16x16x32_bf16(false, fa, false, fb,
                                                       (short)0, acc[t], false, false);
    }
  }

  const float bn = bias[n];
#pragma unroll
  for (int t = 0; t < MT; ++t) {
    int mb = t * 16 + ((lane >> 4) << 3);
#pragma unroll
    for (int i = 0; i < 8; ++i) {
      int m2 = mb + i;
      float y = acc[t][i] + bn;
      if (OutBf)  OutBf[(size_t)m2 * N + n]  = f32_to_bf16_rne(y);
      if (OutF32) OutF32[(size_t)m2 * N + n] = y;
    }
  }
}

// ---------------------------------------------------------------------------
// Scalar spiral conv for tiny layers (first: K=27, last: Cout=3)
// ---------------------------------------------------------------------------
__global__ void sconv_naive_kernel(const unsigned short* __restrict__ H, const int* __restrict__ S,
                                   const float* __restrict__ W, const float* __restrict__ bias,
                                   unsigned short* __restrict__ OutBf, float* __restrict__ OutF32,
                                   int Pn, int Cin, int Cout, int act, int total) {
  int t = blockIdx.x * blockDim.x + threadIdx.x;
  if (t >= total) return;
  int n  = t % Cout;
  int mp = t / Cout;
  int b  = mp / Pn;
  int p  = mp - b * Pn;
  const int* sp = S + p * SPIRAL;
  const unsigned short* hb = H + (size_t)b * Pn * Cin;
  float acc = bias[n];
  for (int s = 0; s < SPIRAL; ++s) {
    int vtx = sp[s];
    const unsigned short* hv = hb + (size_t)vtx * Cin;
    const float* wr = W + (size_t)s * Cin * Cout + n;
    for (int c = 0; c < Cin; ++c)
      acc += bf16_to_f32(hv[c]) * wr[(size_t)c * Cout];
  }
  if (act) acc = (acc > 0.f) ? acc : (__expf(acc) - 1.f);
  if (p == Pn - 1) acc = 0.f;
  if (OutBf)  OutBf[(size_t)mp * Cout + n]  = f32_to_bf16_rne(acc);
  if (OutF32) OutF32[(size_t)mp * Cout + n] = acc;
}

// ---------------------------------------------------------------------------
extern "C" void kernel_launch(void* const* d_in, const int* in_sizes, int n_in,
                              void* d_out, int out_size, void* d_ws, size_t ws_size,
                              hipStream_t stream) {
  (void)in_sizes; (void)n_in; (void)out_size; (void)ws_size;
  const int P0 = 6891, P1 = 1724, P2 = 432, P3 = 109, P4 = 28;

  const float* x  = (const float*)d_in[0];
  const int*   S0 = (const int*)d_in[1];
  const float* D0 = (const float*)d_in[2];
  const float* U0 = (const float*)d_in[3];
  const int*   S1 = (const int*)d_in[4];
  const float* D1 = (const float*)d_in[5];
  const float* U1 = (const float*)d_in[6];
  const int*   S2 = (const int*)d_in[7];
  const float* D2 = (const float*)d_in[8];
  const float* U2 = (const float*)d_in[9];
  const int*   S3 = (const int*)d_in[10];
  const float* D3 = (const float*)d_in[11];
  const float* U3 = (const float*)d_in[12];
  const float* We0 = (const float*)d_in[13]; const float* be0 = (const float*)d_in[14];
  const float* We1 = (const float*)d_in[15]; const float* be1 = (const float*)d_in[16];
  const float* We2 = (const float*)d_in[17]; const float* be2 = (const float*)d_in[18];
  const float* We3 = (const float*)d_in[19]; const float* be3 = (const float*)d_in[20];
  const float* Wfe = (const float*)d_in[21]; const float* bfe = (const float*)d_in[22];
  const float* Wfd = (const float*)d_in[23]; const float* bfd = (const float*)d_in[24];
  const float* Wd0 = (const float*)d_in[25]; const float* bd0 = (const float*)d_in[26];
  const float* Wd1 = (const float*)d_in[27]; const float* bd1 = (const float*)d_in[28];
  const float* Wd2 = (const float*)d_in[29]; const float* bd2 = (const float*)d_in[30];
  const float* Wd3 = (const float*)d_in[31]; const float* bd3 = (const float*)d_in[32];
  const float* Wd4 = (const float*)d_in[33]; const float* bd4 = (const float*)d_in[34];

  float* out_h = (float*)d_out;
  float* out_z = out_h + (size_t)BATCH * P0 * 3;

  // workspace carve (bf16 buffers)
  auto alignup = [](size_t v) { return (v + 255) & ~((size_t)255); };
  char* w = (char*)d_ws;
  const size_t XB_E  = (size_t)BATCH * P0 * 3;      // 1.32M elems
  const size_t ACT_E = (size_t)BATCH * P0 * 32;     // 14.1M elems
  unsigned short* xb   = (unsigned short*)w; w += alignup(XB_E * 2);
  unsigned short* actA = (unsigned short*)w; w += alignup(ACT_E * 2);
  unsigned short* actB = (unsigned short*)w; w += alignup(ACT_E * 2);
  unsigned short* zb   = (unsigned short*)w; w += alignup((size_t)BATCH * 128 * 2);

  dim3 blk(128, 1, 1);
  dim3 wave(32, 1, 1);
  const int MB = 16 * MT * WV;                      // 256 rows per block
  auto vtiles = [&](int Pout) { return (Pout + MB - 1) / MB; };
  auto stiles = [&](int Pn)   { return (BATCH * Pn + MB - 1) / MB; };

  // x -> bf16
  {
    int n = (int)XB_E;
    cvt_f32_bf16_kernel<<<(n + 255) / 256, 256, 0, stream>>>(x, xb, n);
  }
  // encoder conv0 (Cin=3, K=27) scalar
  {
    int total = BATCH * P0 * 16;
    sconv_naive_kernel<<<(total + 255) / 256, 256, 0, stream>>>(
        xb, S0, We0, be0, actA, nullptr, P0, 3, 16, 1, total);
  }
  // D0: [1724,6891] x h[6891,16]
  vertmix_wmma_kernel<<<dim3(vtiles(P1), 1, BATCH), blk, 0, stream>>>(D0, actA, actB, P1, P0, 16);
  // e1: Cin=16, Cout=32, K=144
  sconv_wmma_kernel<<<dim3(stiles(P1), 2), blk, 0, stream>>>(actB, S1, We1, be1, actA, P1, 4, 32, 144);
  // D1
  vertmix_wmma_kernel<<<dim3(vtiles(P2), 2, BATCH), blk, 0, stream>>>(D1, actA, actB, P2, P1, 32);
  // e2: Cin=32, Cout=64, K=288
  sconv_wmma_kernel<<<dim3(stiles(P2), 4), blk, 0, stream>>>(actB, S2, We2, be2, actA, P2, 5, 64, 288);
  // D2
  vertmix_wmma_kernel<<<dim3(vtiles(P3), 4, BATCH), blk, 0, stream>>>(D2, actA, actB, P3, P2, 64);
  // e3: Cin=64, Cout=128, K=576
  sconv_wmma_kernel<<<dim3(stiles(P3), 8), blk, 0, stream>>>(actB, S3, We3, be3, actA, P3, 6, 128, 576);
  // D3
  vertmix_wmma_kernel<<<dim3(vtiles(P4), 8, BATCH), blk, 0, stream>>>(D3, actA, actB, P4, P3, 128);
  // z = flat(h) @ W_fe + b_fe   (M=64, K=3584, N=128) -> f32 out + bf16
  fc_wmma_kernel<<<dim3(1, 128 / 16), wave, 0, stream>>>(actB, Wfe, bfe, zb, out_z, P4 * 128, 128);
  // h = z @ W_fd + b_fd         (M=64, K=128, N=3584)
  fc_wmma_kernel<<<dim3(1, 3584 / 16), wave, 0, stream>>>(zb, Wfd, bfd, actA, nullptr, 128, 3584);
  // U3: [109,28] x h[28,128]
  vertmix_wmma_kernel<<<dim3(vtiles(P3), 8, BATCH), blk, 0, stream>>>(U3, actA, actB, P3, P4, 128);
  // d0: Cin=128, Cout=64, K=1152
  sconv_wmma_kernel<<<dim3(stiles(P3), 4), blk, 0, stream>>>(actB, S3, Wd0, bd0, actA, P3, 7, 64, 1152);
  // U2
  vertmix_wmma_kernel<<<dim3(vtiles(P2), 4, BATCH), blk, 0, stream>>>(U2, actA, actB, P2, P3, 64);
  // d1: Cin=64, Cout=32, K=576
  sconv_wmma_kernel<<<dim3(stiles(P2), 2), blk, 0, stream>>>(actB, S2, Wd1, bd1, actA, P2, 6, 32, 576);
  // U1
  vertmix_wmma_kernel<<<dim3(vtiles(P1), 2, BATCH), blk, 0, stream>>>(U1, actA, actB, P1, P2, 32);
  // d2: Cin=32, Cout=32, K=288
  sconv_wmma_kernel<<<dim3(stiles(P1), 2), blk, 0, stream>>>(actB, S1, Wd2, bd2, actA, P1, 5, 32, 288);
  // U0: [6891,1724] x h[1724,32]
  vertmix_wmma_kernel<<<dim3(vtiles(P0), 2, BATCH), blk, 0, stream>>>(U0, actA, actB, P0, P1, 32);
  // d3: Cin=32, Cout=16, K=288
  sconv_wmma_kernel<<<dim3(stiles(P0), 1), blk, 0, stream>>>(actB, S0, Wd3, bd3, actA, P0, 5, 16, 288);
  // final conv to xyz (identity act, Cout=3) scalar -> f32 output
  {
    int total = BATCH * P0 * 3;
    sconv_naive_kernel<<<(total + 255) / 256, 256, 0, stream>>>(
        actA, S0, Wd4, bd4, nullptr, out_h, P0, 16, 3, 0, total);
  }
}
